// AttentionBiLSTM_Seqence_16947940950744
// MI455X (gfx1250) — matrix operand order repaired and checked by reading, not compiled, not run on hardware
//
#include <hip/hip_runtime.h>

// ---------------- problem constants ----------------
#define S_LEN 512
#define BATCH 64
#define EMB   512
#define HID   768
#define G4H   3072   // 4*HID
#define NLBL  32
#define PAD_WORD 1
#define PAD_BIAS 10000.0f

// ---------------- WMMA types (CDNA5 gfx1250, wave32) ----------------
typedef __attribute__((ext_vector_type(16))) __bf16 v16bf;
typedef __attribute__((ext_vector_type(8)))  float  v8f;

union ABfrag { uint4 q[2]; v16bf v; };   // 32 bytes: one A or B fragment per lane

__device__ __forceinline__ unsigned short f2bf(float f) {
  unsigned int u = __float_as_uint(f);
  return (unsigned short)((u + 0x7FFFu + ((u >> 16) & 1u)) >> 16);  // RNE
}
__device__ __forceinline__ float sigm(float x) { return 1.0f / (1.0f + __expf(-x)); }

// ======================================================================
// Register-blocked GEMM: C[M,N] = A[M,K](bf16,row-major) @ Bt[N,K](bf16,N-major) + bias
// One 32x32 macro-tile per wave (2x2 WMMA accumulators), 4 waves/block.
// Per k-step: 4 b128-pair loads feed 4 v_wmma_f32_16x16x32_bf16 (2x the
// FLOP/byte of the 1-tile version).  Requires M%32==0, N%32==0, K%32==0.
// ======================================================================
__global__ __launch_bounds__(128)
void gemm_bf16_tn(const unsigned short* __restrict__ A,
                  const unsigned short* __restrict__ Bt,
                  const float* __restrict__ bias,
                  float* __restrict__ C,
                  int M, int N, int K)
{
  const int wave = threadIdx.x >> 5;
  const int lane = threadIdx.x & 31;
  const int tile = blockIdx.x * 4 + wave;          // macro-tile id
  const int ntN  = N >> 5;                         // macro-tiles along N
  if (tile >= (M >> 5) * ntN) return;
  const int mt = tile / ntN;
  const int nt = tile - mt * ntN;
  const int l16 = lane & 15;
  const int hh  = lane >> 4;
  // A lane layout (16b A 16x32): elems 0-7 -> K = k + hh*8 .. +7 ; elems 8-15 -> +16
  const unsigned short* pa0 = A  + (size_t)(mt * 32 + l16) * K + hh * 8;
  const unsigned short* pa1 = pa0 + (size_t)16 * K;
  // B lane layout (16b B 32x16): elems 0-7 -> K = k + hh*8 .. +7 ; elems 8-15 -> +16, column l16
  const unsigned short* pb0 = Bt + (size_t)(nt * 32 + l16) * K + hh * 8;
  const unsigned short* pb1 = pb0 + (size_t)16 * K;
  v8f acc00 = {}, acc01 = {}, acc10 = {}, acc11 = {};
  for (int k = 0; k < K; k += 32) {
    ABfrag a0, a1, b0, b1;
    a0.q[0] = *(const uint4*)(pa0 + k);
    a0.q[1] = *(const uint4*)(pa0 + k + 16);
    a1.q[0] = *(const uint4*)(pa1 + k);
    a1.q[1] = *(const uint4*)(pa1 + k + 16);
    b0.q[0] = *(const uint4*)(pb0 + k);
    b0.q[1] = *(const uint4*)(pb0 + k + 16);
    b1.q[0] = *(const uint4*)(pb1 + k);
    b1.q[1] = *(const uint4*)(pb1 + k + 16);
    __builtin_prefetch(pa0 + k + 128, 0, 0);   // global_prefetch_b8
    __builtin_prefetch(pa1 + k + 128, 0, 0);
    __builtin_prefetch(pb0 + k + 128, 0, 0);
    __builtin_prefetch(pb1 + k + 128, 0, 0);
    acc00 = __builtin_amdgcn_wmma_f32_16x16x32_bf16(false, a0.v, false, b0.v,
                                                    (short)0, acc00, false, false);
    acc01 = __builtin_amdgcn_wmma_f32_16x16x32_bf16(false, a0.v, false, b1.v,
                                                    (short)0, acc01, false, false);
    acc10 = __builtin_amdgcn_wmma_f32_16x16x32_bf16(false, a1.v, false, b0.v,
                                                    (short)0, acc10, false, false);
    acc11 = __builtin_amdgcn_wmma_f32_16x16x32_bf16(false, a1.v, false, b1.v,
                                                    (short)0, acc11, false, false);
  }
  const int col0 = nt * 32 + l16;
  const int col1 = col0 + 16;
  const float bv0 = bias ? bias[col0] : 0.0f;
  const float bv1 = bias ? bias[col1] : 0.0f;
  // C/D layout: lane -> n = lane%16 ; m = 8*(lane/16) + vgpr_index
  float* r0 = C + (size_t)(mt * 32 + hh * 8) * N;
  float* r1 = C + (size_t)(mt * 32 + 16 + hh * 8) * N;
  #pragma unroll
  for (int i = 0; i < 8; ++i) {
    r0[(size_t)i * N + col0] = acc00[i] + bv0;
    r0[(size_t)i * N + col1] = acc01[i] + bv1;
    r1[(size_t)i * N + col0] = acc10[i] + bv0;
    r1[(size_t)i * N + col1] = acc11[i] + bv1;
  }
}

// ======================================================================
// One LSTM forward time-step, fully fused:
//   g = xf[t] + hid_in @ Wh_f   (biases pre-folded into xf)
//   c = sig(f)*c + sig(r)*tanh(g) ; hid = sig(o)*tanh(c)
// Block = (m-tile, h-tile); wave w computes gate w's 16x16 tile; LDS exchange.
// Kept at 1 tile/wave: only 768 tiles of work per step -> occupancy first.
// ======================================================================
__global__ __launch_bounds__(128)
void lstm_fwd_step(const float* __restrict__ xf_t,          // [64][3072]
                   const unsigned short* __restrict__ WhT,  // [3072][768] bf16
                   const unsigned short* __restrict__ hid_in,  // [64][768] bf16
                   unsigned short* __restrict__ hid_out,       // [64][768] bf16
                   float* __restrict__ cst,                    // [64][768] f32
                   unsigned short* __restrict__ hs_t)          // [64][1536] bf16 (cols 0..767)
{
  __shared__ float lds[4][8][32];
  const int wave = threadIdx.x >> 5, lane = threadIdx.x & 31;
  const int mt = blockIdx.x, ht = blockIdx.y;
  const int l16 = lane & 15, hh = lane >> 4;
  const unsigned short* pa = hid_in + (size_t)(mt * 16 + l16) * HID + hh * 8;
  const unsigned short* pb = WhT + (size_t)(wave * HID + ht * 16 + l16) * HID + hh * 8;
  v8f acc = {};
  #pragma unroll 4
  for (int k = 0; k < HID; k += 32) {
    ABfrag a, b;
    a.q[0] = *(const uint4*)(pa + k);
    a.q[1] = *(const uint4*)(pa + k + 16);
    b.q[0] = *(const uint4*)(pb + k);
    b.q[1] = *(const uint4*)(pb + k + 16);
    acc = __builtin_amdgcn_wmma_f32_16x16x32_bf16(false, a.v, false, b.v,
                                                  (short)0, acc, false, false);
  }
  #pragma unroll
  for (int i = 0; i < 8; ++i) lds[wave][i][lane] = acc[i];
  __syncthreads();
  // 256 tile elements, 128 threads -> 2 each
  #pragma unroll
  for (int u = 0; u < 2; ++u) {
    const int e = threadIdx.x * 2 + u;
    const int l = e & 31, i = e >> 5;
    const int m = ((l >> 4) << 3) + i;     // C/D layout: m = 8*(lane>=16) + vgpr
    const int n = l & 15;
    const int b = mt * 16 + m, h = ht * 16 + n;
    const float* x = xf_t + (size_t)b * G4H + h;
    const float gr = lds[0][i][l] + x[0];
    const float gf = lds[1][i][l] + x[HID];
    const float gg = lds[2][i][l] + x[2 * HID];
    const float go = lds[3][i][l] + x[3 * HID];
    const float cn = sigm(gf) * cst[b * HID + h] + sigm(gr) * tanhf(gg);
    const float hn = sigm(go) * tanhf(cn);
    cst[b * HID + h] = cn;
    const unsigned short hb = f2bf(hn);
    hid_out[b * HID + h] = hb;
    hs_t[(size_t)b * (2 * HID) + h] = hb;
  }
}

// ======================================================================
// Backward direction: gates are precomputed (gb + hT@Wh_b); pure elementwise
// reverse scan, one thread per (b,h), sequential over s. Bandwidth bound.
// ======================================================================
__global__ __launch_bounds__(256)
void bwd_scan(const float* __restrict__ gb,   // [S][64][3072]
              const float* __restrict__ hc,   // [64][3072]  (hT @ Wh_b)
              unsigned short* __restrict__ hs) // [S][64][1536] (writes cols 768..1535)
{
  const int tid = blockIdx.x * 256 + threadIdx.x;     // 0..49151
  const int b = tid / HID, h = tid - b * HID;
  const float hr = hc[b * G4H + h];
  const float hf = hc[b * G4H + HID + h];
  const float hg = hc[b * G4H + 2 * HID + h];
  const float ho = hc[b * G4H + 3 * HID + h];
  float c2 = 0.0f;
  for (int s = S_LEN - 1; s >= 0; --s) {
    const float* g = gb + ((size_t)s * BATCH + b) * G4H + h;
    const float r  = sigm(g[0] + hr);
    const float f  = sigm(g[HID] + hf);
    const float gg = tanhf(g[2 * HID] + hg);
    const float o  = sigm(g[3 * HID] + ho);
    c2 = f * c2 + r * gg;
    hs[((size_t)s * BATCH + b) * (2 * HID) + HID + h] = f2bf(o * tanhf(c2));
  }
}

// ======================================================================
// out[S*B,32] = hs[S*B,1536] @ Wout + bout, with pad bias fused on column 0.
// One wave covers one M-tile x BOTH N-tiles (L=32): hs streamed once.
// ======================================================================
__global__ __launch_bounds__(128)
void out_gemm(const unsigned short* __restrict__ hs,   // [32768][1536] bf16
              const unsigned short* __restrict__ WoT,  // [32][1536] bf16
              const float* __restrict__ bout,
              const int* __restrict__ tokens,          // [S*B]
              float* __restrict__ out)                 // [32768][32]
{
  const int wave = threadIdx.x >> 5, lane = threadIdx.x & 31;
  const int mt = blockIdx.x * 4 + wave;                // 0..2047
  const int l16 = lane & 15, hh = lane >> 4;
  const unsigned short* pa  = hs  + (size_t)(mt * 16 + l16) * (2 * HID) + hh * 8;
  const unsigned short* pb0 = WoT + (size_t)l16 * (2 * HID) + hh * 8;
  const unsigned short* pb1 = pb0 + (size_t)16 * (2 * HID);
  v8f acc0 = {}, acc1 = {};
  #pragma unroll 4
  for (int k = 0; k < 2 * HID; k += 32) {
    ABfrag a, b0, b1;
    a.q[0]  = *(const uint4*)(pa + k);
    a.q[1]  = *(const uint4*)(pa + k + 16);
    b0.q[0] = *(const uint4*)(pb0 + k);
    b0.q[1] = *(const uint4*)(pb0 + k + 16);
    b1.q[0] = *(const uint4*)(pb1 + k);
    b1.q[1] = *(const uint4*)(pb1 + k + 16);
    acc0 = __builtin_amdgcn_wmma_f32_16x16x32_bf16(false, a.v, false, b0.v,
                                                   (short)0, acc0, false, false);
    acc1 = __builtin_amdgcn_wmma_f32_16x16x32_bf16(false, a.v, false, b1.v,
                                                   (short)0, acc1, false, false);
  }
  const float bv0 = bout[l16];
  const float bv1 = bout[l16 + 16];
  #pragma unroll
  for (int i = 0; i < 8; ++i) {
    const int row = mt * 16 + hh * 8 + i;
    float v0 = acc0[i] + bv0;
    if (l16 == 0 && tokens[row] == PAD_WORD) v0 += PAD_BIAS;
    out[(size_t)row * NLBL + l16]      = v0;
    out[(size_t)row * NLBL + l16 + 16] = acc1[i] + bv1;
  }
}

// ---------------- prep kernels ----------------
__global__ __launch_bounds__(256)
void convert_transpose(const float* __restrict__ W,      // [K][N] f32
                       unsigned short* __restrict__ Wt,  // [N][K] bf16
                       int K, int N)
{
  const int idx = blockIdx.x * 256 + threadIdx.x;
  if (idx >= N * K) return;
  const int n = idx / K, k = idx - n * K;
  Wt[idx] = f2bf(W[(size_t)k * N + n]);
}

__global__ __launch_bounds__(256)
void add_vec(const float* __restrict__ a, const float* __restrict__ b,
             float* __restrict__ o, int n)
{
  const int i = blockIdx.x * 256 + threadIdx.x;
  if (i < n) o[i] = a[i] + b[i];
}

__global__ __launch_bounds__(256)
void gather_embed(const int* __restrict__ tokens,       // [S*B]
                  const float* __restrict__ emb,        // [V][E]
                  unsigned short* __restrict__ out)     // [S*B][E] bf16
{
  const size_t idx = (size_t)blockIdx.x * 256 + threadIdx.x;
  const int e = (int)(idx & (EMB - 1));
  const size_t sb = idx >> 9;                           // EMB == 512
  out[idx] = f2bf(emb[(size_t)tokens[sb] * EMB + e]);
}

__global__ __launch_bounds__(256)
void zero_state(float* __restrict__ c, unsigned short* __restrict__ h0,
                unsigned short* __restrict__ h1)
{
  const int i = blockIdx.x * 256 + threadIdx.x;
  if (i < BATCH * HID) { c[i] = 0.0f; h0[i] = 0; h1[i] = 0; }
}

// ======================================================================
// host launcher
// ======================================================================
extern "C" void kernel_launch(void* const* d_in, const int* in_sizes, int n_in,
                              void* d_out, int out_size, void* d_ws, size_t ws_size,
                              hipStream_t stream)
{
  (void)in_sizes; (void)n_in; (void)out_size; (void)ws_size;
  const int*   tokens = (const int*)  d_in[0];
  // d_in[1] = tags (unused by forward)
  const float* emb    = (const float*)d_in[2];
  const float* Wi_f   = (const float*)d_in[3];
  const float* bi_f   = (const float*)d_in[4];
  const float* Wh_f   = (const float*)d_in[5];
  const float* bh_f   = (const float*)d_in[6];
  const float* Wi_b   = (const float*)d_in[7];
  const float* bi_b   = (const float*)d_in[8];
  const float* Wh_b   = (const float*)d_in[9];
  const float* bh_b   = (const float*)d_in[10];
  const float* Wout   = (const float*)d_in[11];
  const float* bout   = (const float*)d_in[12];
  float* out = (float*)d_out;

  char* base = (char*)d_ws;
  size_t off = 0;
  auto carve = [&](size_t bytes) -> void* {
    void* p = base + off;
    off = (off + bytes + 255) & ~(size_t)255;
    return p;
  };

  unsigned short* embB = (unsigned short*)carve((size_t)S_LEN * BATCH * EMB * 2);   // 32 MB
  float*          xf   = (float*)         carve((size_t)S_LEN * BATCH * G4H * 4);  // 384 MB
  float*          gbuf = (float*)         carve((size_t)S_LEN * BATCH * G4H * 4);  // 384 MB
  unsigned short* hs   = (unsigned short*)carve((size_t)S_LEN * BATCH * 2 * HID * 2); // 96 MB
  unsigned short* WiTf = (unsigned short*)carve((size_t)EMB * G4H * 2);
  unsigned short* WiTb = (unsigned short*)carve((size_t)EMB * G4H * 2);
  unsigned short* WhTf = (unsigned short*)carve((size_t)HID * G4H * 2);
  unsigned short* WhTb = (unsigned short*)carve((size_t)HID * G4H * 2);
  unsigned short* WoT  = (unsigned short*)carve((size_t)2 * HID * NLBL * 2);
  float* bsf  = (float*)carve(G4H * 4);
  float* bsb  = (float*)carve(G4H * 4);
  unsigned short* hid0 = (unsigned short*)carve((size_t)BATCH * HID * 2);
  unsigned short* hid1 = (unsigned short*)carve((size_t)BATCH * HID * 2);
  float* cst = (float*)carve((size_t)BATCH * HID * 4);
  float* hc  = (float*)carve((size_t)BATCH * G4H * 4);

  // ---- weight prep: bf16 + transpose to N-major, fold biases ----
  convert_transpose<<<(EMB * G4H + 255) / 256, 256, 0, stream>>>(Wi_f, WiTf, EMB, G4H);
  convert_transpose<<<(EMB * G4H + 255) / 256, 256, 0, stream>>>(Wi_b, WiTb, EMB, G4H);
  convert_transpose<<<(HID * G4H + 255) / 256, 256, 0, stream>>>(Wh_f, WhTf, HID, G4H);
  convert_transpose<<<(HID * G4H + 255) / 256, 256, 0, stream>>>(Wh_b, WhTb, HID, G4H);
  convert_transpose<<<(2 * HID * NLBL + 255) / 256, 256, 0, stream>>>(Wout, WoT, 2 * HID, NLBL);
  add_vec<<<(G4H + 255) / 256, 256, 0, stream>>>(bi_f, bh_f, bsf, G4H);
  add_vec<<<(G4H + 255) / 256, 256, 0, stream>>>(bi_b, bh_b, bsb, G4H);

  // ---- embedding gather (f32 -> bf16) ----
  gather_embed<<<(S_LEN * BATCH * EMB) / 256, 256, 0, stream>>>(tokens, emb, embB);
  zero_state<<<(BATCH * HID + 255) / 256, 256, 0, stream>>>(cst, hid0, hid1);

  // ---- parallel input-gate GEMMs: xf / gb = emb @ Wi + (bi + bh) ----
  const int tilesBig = (S_LEN * BATCH / 32) * (G4H / 32);  // 98304 macro-tiles
  gemm_bf16_tn<<<tilesBig / 4, 128, 0, stream>>>(embB, WiTf, bsf, xf,
                                                 S_LEN * BATCH, G4H, EMB);
  gemm_bf16_tn<<<tilesBig / 4, 128, 0, stream>>>(embB, WiTb, bsb, gbuf,
                                                 S_LEN * BATCH, G4H, EMB);

  // ---- forward recurrence: 512 fused WMMA steps, double-buffered hidden ----
  const unsigned short* hin = hid0;
  unsigned short*       hout = hid1;
  for (int t = 0; t < S_LEN; ++t) {
    lstm_fwd_step<<<dim3(BATCH / 16, HID / 16), 128, 0, stream>>>(
        xf + (size_t)t * BATCH * G4H, WhTf, hin, hout, cst,
        hs + (size_t)t * BATCH * 2 * HID);
    unsigned short* tmp = (unsigned short*)hin;
    hin = hout; hout = tmp;
  }
  // hin == hT (final forward hidden)

  // ---- hc = hT @ Wh_b (constant hidden contribution for backward dir) ----
  const int tilesHC = (BATCH / 32) * (G4H / 32);           // 192 macro-tiles
  gemm_bf16_tn<<<tilesHC / 4, 128, 0, stream>>>(hin, WhTb, nullptr, hc,
                                                BATCH, G4H, HID);

  // ---- backward elementwise reverse scan ----
  bwd_scan<<<(BATCH * HID) / 256, 256, 0, stream>>>(gbuf, hc, hs);

  // ---- output projection + pad bias ----
  out_gemm<<<(S_LEN * BATCH / 16) / 4, 128, 0, stream>>>(hs, WoT, bout, tokens, out);
}
